// ForceOnlyDynamicsDecoderRNN_25606595019229
// MI455X (gfx1250) — matrix operand order repaired and checked by reading, not compiled, not run on hardware
//
#include <hip/hip_runtime.h>
#include <hip/hip_bf16.h>

// ---------------------------------------------------------------------------
// GRU recurrence, B=1024, T=256, H=512, GRU_IN=38 (f:2, y:4, z:32), OUT=4.
//
// Strategy (MI455X / gfx1250, wave32, WMMA):
//  * Persistent kernel: 32 workgroups x 1024 threads; WG g owns batch rows
//    [32g, 32g+32).  Wave w (0..31) owns hidden columns [16w, 16w+16) for all
//    3 gates and both 16-row M-tiles.
//  * h-GEMM per step via v_wmma_f32_16x16x32_f16, K-loop of 16 (K=512),
//    r/z/n B fragments double-buffered (one k-iter of L2 loads in flight).
//    Main K-loop is branch-free (spill-free allocation).
//    h kept in LDS twice: f16 (WMMA A) + f32 (exact update).
//  * x-GEMM split: z_dyn part + biases precomputed once into `giz` (f32,
//    B x 1536) and streamed from L2 in the gate phase; f_t/y_prev are a
//    rank-6 VALU update.
//  * W_hh pre-packed into WMMA-B fragment order (f16): per (n-tile q, k-iter)
//    fragment = 32 lanes x 16 halves = 1KB, read fully coalesced from L2
//    each step (W stream is L2-resident).
//  * y_t = h_new @ W_out.T: the packed W_out B-block (16KB) is copied to LDS
//    once; waves 0/1 run a short all-LDS y K-loop after the main K-loop
//    (16 WMMAs fed by ds_load_b128 only), then publish y to LDS ybuf.
// ---------------------------------------------------------------------------

typedef __attribute__((ext_vector_type(16))) _Float16 v16h;
typedef __attribute__((ext_vector_type(8)))  float    v8f;

union FragU { uint4 u[2]; v16h v; };   // 32B: one f16 WMMA A/B operand per lane

#define T_STEPS 256
#define BT      32          // batch rows per workgroup
#define HS16    520         // f16 h LDS stride (halves): 520*2B -> lane banks 4l..4l+3
#define HS32    516         // f32 h LDS stride (floats)
#define QSTRIDE 16384       // bytes per n-tile in packed W: 16 k-iters * 1KB
#define WP_ELEMS (97*16*32*16)          // 97 n-tiles x 16 k-iters x 32 lanes x 16 halves
#define WP_BYTES (WP_ELEMS*2)
#define SMEM_H16   0
#define SMEM_H32   (BT*HS16*2)                       // 33280
#define SMEM_FALL  (SMEM_H32 + BT*HS32*4)            // 99328
#define SMEM_YBUF  (SMEM_FALL + BT*T_STEPS*2*4)      // 164864
#define SMEM_WY    (SMEM_YBUF + BT*4*4)              // 165376 (16KB W_out B-block)
#define SMEM_TOTAL (SMEM_WY + QSTRIDE)               // 181760 bytes (<320KB WGP LDS)

__device__ __forceinline__ v8f wmma_f16(v16h a, v16h b, v8f c) {
    return __builtin_amdgcn_wmma_f32_16x16x32_f16(
        /*neg_a=*/false, a, /*neg_b=*/false, b,
        /*c_mod=*/(short)0, c, /*reuse_a=*/false, /*reuse_b=*/false);
}

__device__ __forceinline__ float sigmoidf_(float x) {
    return 1.0f / (1.0f + __expf(-x));
}

// ---------------------------------------------------------------------------
// Prep kernel 1: pack W_hh (1536x512 f32) and W_out (4x512 f32) into f16
// WMMA-B fragment order.  Fragment (q,k): lane l, half h holds
//   B[K = k*32 + (h<8?0:16) + (l<16?0:8) + (h&7)][N = base_n(q) + (l&15)]
// matching the ISA's wave32 16-bit B operand layout (2 K per VGPR).
// q = gate*32 + hidden_tile for q<96;  q==96 is W_out (cols 0..3, rest 0).
// ---------------------------------------------------------------------------
__global__ void pack_w_kernel(const float* __restrict__ W_hh,
                              const float* __restrict__ W_out,
                              _Float16* __restrict__ Wp) {
    int idx = blockIdx.x * 256 + threadIdx.x;
    if (idx >= WP_ELEMS) return;
    int hh = idx & 15;
    int l  = (idx >> 4) & 31;
    int k  = (idx >> 9) & 15;
    int q  = idx >> 13;
    int kg = k * 32 + ((hh & 8) ? 16 : 0) + ((l & 16) ? 8 : 0) + (hh & 7);
    float v;
    if (q < 96) {
        int gate = q >> 5, ht = q & 31;
        int n = gate * 512 + ht * 16 + (l & 15);
        v = W_hh[(size_t)n * 512 + kg];
    } else {
        int o = l & 15;
        v = (o < 4) ? W_out[(size_t)o * 512 + kg] : 0.0f;
    }
    Wp[idx] = (_Float16)v;
}

// ---------------------------------------------------------------------------
// Prep kernel 2: giz[b][gate*512+j] = z_dyn[b] . W_ih[gate*512+j][6:38]
//                + b_ih[gate*512+j] + (gate<2 ? b_hh[gate*512+j] : 0)
// (for the n-gate, b_hh_n seeds the gh accumulator instead, since
//  n = tanh(i_n + r * h_n) keeps i and h parts separate).
// ---------------------------------------------------------------------------
__global__ void giz_kernel(const float* __restrict__ z_dyn,
                           const float* __restrict__ W_ih,
                           const float* __restrict__ b_ih,
                           const float* __restrict__ b_hh,
                           float* __restrict__ giz) {
    int idx = blockIdx.x * 256 + threadIdx.x;
    if (idx >= 1024 * 1536) return;
    int b  = idx / 1536;
    int cc = idx - b * 1536;
    int gate = cc >> 9;
    const float* zr = z_dyn + (size_t)b * 32;
    const float* wr = W_ih + (size_t)cc * 38 + 6;
    float acc = b_ih[cc] + (gate < 2 ? b_hh[cc] : 0.0f);
#pragma unroll 8
    for (int d = 0; d < 32; ++d) acc += zr[d] * wr[d];
    giz[idx] = acc;
}

// ---------------------------------------------------------------------------
// Main persistent GRU kernel.
// ---------------------------------------------------------------------------
__global__ __launch_bounds__(1024)
void gru_main_kernel(const float* __restrict__ f_seq,
                     const float* __restrict__ init_y,
                     const float* __restrict__ W_ih,
                     const float* __restrict__ b_hh,
                     const float* __restrict__ b_out,
                     const float* __restrict__ W_out,
                     const _Float16* __restrict__ Wp,
                     const float* __restrict__ giz,
                     float* __restrict__ out) {
    extern __shared__ char smem[];
    _Float16* h16 = (_Float16*)(smem + SMEM_H16);   // [BT][HS16] f16 h (WMMA A)
    float*    h32 = (float*)(smem + SMEM_H32);      // [BT][HS32] f32 h (exact)
    float*    fall = (float*)(smem + SMEM_FALL);    // [BT][T][2] f_seq tile
    float*    ybuf = (float*)(smem + SMEM_YBUF);    // [BT][4]    y_prev
    char*     wy   = smem + SMEM_WY;                // 16KB W_out B fragments

    const int tid = threadIdx.x;
    const int g   = blockIdx.x;

    // ---- prologue: zero h, stage f_seq tile + W_out B-block, load init_y ----
    for (int i = tid; i < BT * HS16; i += 1024) h16[i] = (_Float16)0.0f;
    for (int i = tid; i < BT * HS32; i += 1024) h32[i] = 0.0f;
    {
        const float4* src4 = (const float4*)(f_seq + (size_t)g * BT * T_STEPS * 2);
        float4* dst4 = (float4*)fall;
        for (int i = tid; i < BT * T_STEPS * 2 / 4; i += 1024) dst4[i] = src4[i];
        // W_out packed block (q=96): 16KB -> LDS, one uint4 per thread
        ((uint4*)wy)[tid] =
            ((const uint4*)((const char*)Wp + (size_t)96 * QSTRIDE))[tid];
    }
    if (tid < BT * 4) ybuf[tid] = init_y[(size_t)g * BT * 4 + tid];
    __syncthreads();

    // ---- per-lane constants ----
    const int swv   = __builtin_amdgcn_readfirstlane(tid >> 5); // scalar wave id
    const int isY   = __builtin_amdgcn_readfirstlane((tid >> 5) < 2 ? 1 : 0);
    const int lane  = tid & 31;
    const int ln    = lane & 15;
    const int halfu = lane >> 4;       // C/D: M = c + 8*halfu (+ 16*mt)
    const int col   = swv * 16 + ln;   // this lane's hidden column (N)

    float w6[3][6];                    // rank-6 input weights for this column
#pragma unroll
    for (int gate = 0; gate < 3; ++gate)
#pragma unroll
        for (int d = 0; d < 6; ++d)
            w6[gate][d] = W_ih[(size_t)(gate * 512 + col) * 38 + d];
    const float bhhn = b_hh[1024 + col];
    const float bo   = (ln < 4) ? b_out[ln] : 0.0f;

    // giz stream base for the gate phase (per-lane, immediate offsets per mt/c)
    const float* gizB = giz + (size_t)(g * BT + halfu * 8) * 1536 + col;

    // loop-invariant stream bases: fragment (q,k) at q*16KB + k*1KB + lane*32B
    const char* wpR = (const char*)Wp + (size_t)swv * QSTRIDE + (size_t)lane * 32;
    const char* wpZ = wpR + (size_t)32 * QSTRIDE;
    const char* wpN = wpR + (size_t)64 * QSTRIDE;
    const char* wyB = wy + (size_t)lane * 32;               // LDS y B-frag base
    const _Float16* aBase  = h16 + ln * HS16 + halfu * 8;   // A-frag base (halves)
    const _Float16* aBaseY = aBase + swv * 16 * HS16;       // y-wave A base (swv<2)

    // ---- time loop ----
    for (int t = 0; t < T_STEPS; ++t) {
        v8f cr[2], cz[2], cn[2];
#pragma unroll
        for (int c = 0; c < 8; ++c) {
            cr[0][c] = 0.0f; cr[1][c] = 0.0f;
            cz[0][c] = 0.0f; cz[1][c] = 0.0f;
            cn[0][c] = bhhn; cn[1][c] = bhhn;
        }

        // r/z/n B-stream double buffers; preload k=0 into slot 0
        FragU fr[2], fz[2], fn[2];
        fr[0].u[0] = *(const uint4*)(wpR);
        fr[0].u[1] = *(const uint4*)(wpR + 16);
        fz[0].u[0] = *(const uint4*)(wpZ);
        fz[0].u[1] = *(const uint4*)(wpZ + 16);
        fn[0].u[0] = *(const uint4*)(wpN);
        fn[0].u[1] = *(const uint4*)(wpN + 16);

        // ---- main K-loop: branch-free, double-buffered ----
#pragma unroll 4
        for (int k = 0; k < 16; ++k) {
            const int cur = k & 1, nxt = cur ^ 1;

            // issue k+1's r/z/n loads before consuming k's (latency overlap)
            if (k < 15) {
                const int offn = (k + 1) * 1024;
                fr[nxt].u[0] = *(const uint4*)(wpR + offn);
                fr[nxt].u[1] = *(const uint4*)(wpR + offn + 16);
                fz[nxt].u[0] = *(const uint4*)(wpZ + offn);
                fz[nxt].u[1] = *(const uint4*)(wpZ + offn + 16);
                fn[nxt].u[0] = *(const uint4*)(wpN + offn);
                fn[nxt].u[1] = *(const uint4*)(wpN + offn + 16);
            }

            // A fragments (h16, both M-tiles), conflict-free LDS b128 loads
            FragU fa0, fa1;
            fa0.u[0] = *(const uint4*)(aBase + k * 32);
            fa0.u[1] = *(const uint4*)(aBase + k * 32 + 16);
            fa1.u[0] = *(const uint4*)(aBase + 16 * HS16 + k * 32);
            fa1.u[1] = *(const uint4*)(aBase + 16 * HS16 + k * 32 + 16);

            cr[0] = wmma_f16(fa0.v, fr[cur].v, cr[0]);
            cr[1] = wmma_f16(fa1.v, fr[cur].v, cr[1]);
            cz[0] = wmma_f16(fa0.v, fz[cur].v, cz[0]);
            cz[1] = wmma_f16(fa1.v, fz[cur].v, cz[1]);
            cn[0] = wmma_f16(fa0.v, fn[cur].v, cn[0]);
            cn[1] = wmma_f16(fa1.v, fn[cur].v, cn[1]);
        }

        // ---- y tail (waves 0/1 only): all-LDS 16-WMMA K-loop ----
        if (isY) {
            v8f cy;
#pragma unroll
            for (int c = 0; c < 8; ++c) cy[c] = 0.0f;
#pragma unroll 4
            for (int k = 0; k < 16; ++k) {
                FragU fay, fy;
                fay.u[0] = *(const uint4*)(aBaseY + k * 32);
                fay.u[1] = *(const uint4*)(aBaseY + k * 32 + 16);
                fy.u[0] = *(const uint4*)(wyB + k * 1024);
                fy.u[1] = *(const uint4*)(wyB + k * 1024 + 16);
                cy = wmma_f16(fay.v, fy.v, cy);
            }
            // publish y_{t-1} = h_{t-1} @ W_out.T + b_out
            if (t > 0 && ln < 4) {
#pragma unroll
                for (int c = 0; c < 8; ++c)
                    ybuf[(swv * 16 + halfu * 8 + c) * 4 + ln] = cy[c] + bo;
            }
        }
        __syncthreads();   // K-loop reads of h16 done; y_prev published

        if (t > 0 && tid < BT * 4) {
            const int m = tid >> 2, o = tid & 3;
            out[((size_t)(g * BT + m) * T_STEPS + (t - 1)) * 4 + o] = ybuf[tid];
        }

        // gates + state update (each wave touches only its own 16 columns);
        // giz (z_dyn part + biases) streamed from L2 with immediate offsets
#pragma unroll
        for (int mt = 0; mt < 2; ++mt)
#pragma unroll
            for (int c = 0; c < 8; ++c) {
                const int m = mt * 16 + halfu * 8 + c;
                const float grv = gizB[(mt * 16 + c) * 1536];
                const float gzv = gizB[(mt * 16 + c) * 1536 + 512];
                const float gnv = gizB[(mt * 16 + c) * 1536 + 1024];
                const float2 f2 = *(const float2*)(fall + m * (T_STEPS * 2) + t * 2);
                const float4 y4 = *(const float4*)(ybuf + m * 4);
                float r6r = w6[0][0]*f2.x + w6[0][1]*f2.y + w6[0][2]*y4.x
                          + w6[0][3]*y4.y + w6[0][4]*y4.z + w6[0][5]*y4.w;
                float r6z = w6[1][0]*f2.x + w6[1][1]*f2.y + w6[1][2]*y4.x
                          + w6[1][3]*y4.y + w6[1][4]*y4.z + w6[1][5]*y4.w;
                float r6n = w6[2][0]*f2.x + w6[2][1]*f2.y + w6[2][2]*y4.x
                          + w6[2][3]*y4.y + w6[2][4]*y4.z + w6[2][5]*y4.w;
                const float rv = sigmoidf_(cr[mt][c] + grv + r6r);
                const float zv = sigmoidf_(cz[mt][c] + gzv + r6z);
                const float nv = tanhf(gnv + r6n + rv * cn[mt][c]);
                const float hold = h32[m * HS32 + col];
                const float hn = (1.0f - zv) * nv + zv * hold;
                h32[m * HS32 + col] = hn;
                h16[m * HS16 + col] = (_Float16)hn;
            }
        __syncthreads();   // h_{t} fully visible for next step's K-loop
    }

    // ---- epilogue: y_{T-1} in exact f32 from LDS h ----
    if (tid < BT * 4) {
        const int m = tid >> 2, o = tid & 3;
        float acc = b_out[o];
        for (int j = 0; j < 512; ++j)
            acc += h32[m * HS32 + j] * W_out[(size_t)o * 512 + j];
        out[((size_t)(g * BT + m) * T_STEPS + (T_STEPS - 1)) * 4 + o] = acc;
    }
}

// ---------------------------------------------------------------------------
extern "C" void kernel_launch(void* const* d_in, const int* in_sizes, int n_in,
                              void* d_out, int out_size, void* d_ws, size_t ws_size,
                              hipStream_t stream) {
    const float* f_seq  = (const float*)d_in[0];
    const float* init_y = (const float*)d_in[1];
    const float* z_dyn  = (const float*)d_in[2];
    const float* W_ih   = (const float*)d_in[3];
    const float* W_hh   = (const float*)d_in[4];
    const float* b_ih   = (const float*)d_in[5];
    const float* b_hh   = (const float*)d_in[6];
    const float* W_out  = (const float*)d_in[7];
    const float* b_out  = (const float*)d_in[8];
    float* out = (float*)d_out;

    _Float16* Wp  = (_Float16*)d_ws;                       // 1,589,248 B
    float*    giz = (float*)((char*)d_ws + WP_BYTES);      // 6,291,456 B

    pack_w_kernel<<<(WP_ELEMS + 255) / 256, 256, 0, stream>>>(W_hh, W_out, Wp);
    giz_kernel<<<(1024 * 1536) / 256, 256, 0, stream>>>(z_dyn, W_ih, b_ih, b_hh, giz);
    gru_main_kernel<<<1024 / BT, 1024, SMEM_TOTAL, stream>>>(
        f_seq, init_y, W_ih, b_hh, b_out, W_out, Wp, giz, out);
}
